// AttentionLayer_26396869001772
// MI455X (gfx1250) — matrix operand order repaired and checked by reading, not compiled, not run on hardware
//
#include <hip/hip_runtime.h>

typedef __bf16 bf16_t;
typedef __attribute__((ext_vector_type(16))) __bf16 v16bf;
typedef __attribute__((ext_vector_type(8)))  __bf16 v8bf;
typedef __attribute__((ext_vector_type(4)))  __bf16 v4bf;
typedef __attribute__((ext_vector_type(8)))  float  v8f;

#define E_DIM   1024
#define S_LEN   2048
#define N_HEADS 16
#define HD      64
#define M_TOK   8192   // B*S = 4*2048

// ---------- CDNA5 async global->LDS copy (16B per lane), ASYNCcnt tracked ----------
__device__ __forceinline__ void async_copy16(unsigned lds_addr, const bf16_t* gaddr) {
  asm volatile("global_load_async_to_lds_b128 %0, %1, off"
               :: "v"(lds_addr), "v"(gaddr) : "memory");
}
__device__ __forceinline__ void wait_async_le2() {
  asm volatile("s_wait_asynccnt 2" ::: "memory");
}
__device__ __forceinline__ void wait_async_0() {
  asm volatile("s_wait_asynccnt 0" ::: "memory");
}
__device__ __forceinline__ void wait_ds_0() {
  asm volatile("s_wait_dscnt 0" ::: "memory");
}

// ---------- WMMA fragment helpers (CDNA5 16x16x32 bf16, wave32) ----------
// A-frag (16x32, M x K): lane (l%16)=row; halves 0..7 -> k=(l/16)*8+0..7,
// halves 8..15 -> k=16+(l/16)*8+0..7   (ISA 7.12.2)
__device__ __forceinline__ v16bf load_a_frag(const bf16_t* base, int row0, int ld, int k0) {
  int lane = threadIdx.x & 31;
  int lm = lane & 15, lh = lane >> 4;
  const bf16_t* p = base + (size_t)(row0 + lm) * ld + k0 + lh * 8;
  v8bf lo = *(const v8bf*)p;
  v8bf hi = *(const v8bf*)(p + 16);
  return __builtin_shufflevector(lo, hi, 0,1,2,3,4,5,6,7,8,9,10,11,12,13,14,15);
}

// B-frag (32x16, K x N) with B[k][n] = src[n][k] (row-major N x K source):
// lane (l%16)=column n; halves 0..7 -> k=(l/16)*8+0..7, 8..15 -> k=16+(l/16)*8+0..7.
__device__ __forceinline__ v16bf load_b_frag(const bf16_t* base, int col0, int ld, int k0) {
  int lane = threadIdx.x & 31;
  int lm = lane & 15, lh = lane >> 4;
  const bf16_t* p = base + (size_t)(col0 + lm) * ld + k0 + lh * 8;
  v8bf lo = *(const v8bf*)p;
  v8bf hi = *(const v8bf*)(p + 16);
  return __builtin_shufflevector(lo, hi, 0,1,2,3,4,5,6,7,8,9,10,11,12,13,14,15);
}

__device__ __forceinline__ v8f wmma_bf16f32(v16bf a, v16bf b, v8f c) {
  return __builtin_amdgcn_wmma_f32_16x16x32_bf16(false, a, false, b, (short)0, c, false, false);
}

// ---------- Kernel 1: LayerNorm (f32) -> bf16 activations ----------
__global__ __launch_bounds__(256)
void ln_cast_kernel(const float* __restrict__ x, const float* __restrict__ gamma,
                    const float* __restrict__ beta, bf16_t* __restrict__ h) {
  int token = blockIdx.x;                       // 8192 tokens
  const float* xr = x + (size_t)token * E_DIM;
  float4 v = ((const float4*)xr)[threadIdx.x];  // 256 threads * 4 = 1024
  float s  = v.x + v.y + v.z + v.w;
  float ss = v.x*v.x + v.y*v.y + v.z*v.z + v.w*v.w;
  for (int m = 16; m >= 1; m >>= 1) {
    s  += __shfl_xor(s,  m, 32);
    ss += __shfl_xor(ss, m, 32);
  }
  __shared__ float red[2][8];
  int wid = threadIdx.x >> 5;
  if ((threadIdx.x & 31) == 0) { red[0][wid] = s; red[1][wid] = ss; }
  __syncthreads();
  float st = 0.f, sst = 0.f;
  #pragma unroll
  for (int i = 0; i < 8; ++i) { st += red[0][i]; sst += red[1][i]; }
  float mu  = st * (1.0f / E_DIM);
  float var = sst * (1.0f / E_DIM) - mu * mu;
  float rs  = rsqrtf(var + 1e-5f);
  float4 gv = ((const float4*)gamma)[threadIdx.x];
  float4 bv = ((const float4*)beta )[threadIdx.x];
  v4bf o;
  o[0] = (bf16_t)((v.x - mu) * rs * gv.x + bv.x);
  o[1] = (bf16_t)((v.y - mu) * rs * gv.y + bv.y);
  o[2] = (bf16_t)((v.z - mu) * rs * gv.z + bv.z);
  o[3] = (bf16_t)((v.w - mu) * rs * gv.w + bv.w);
  *(v4bf*)(h + (size_t)token * E_DIM + threadIdx.x * 4) = o;
}

// ---------- Kernel 2: f32 -> bf16 weight cast ----------
__global__ __launch_bounds__(256)
void cast_w_kernel(const float* __restrict__ src, bf16_t* __restrict__ dst, int n) {
  int i = (blockIdx.x * blockDim.x + threadIdx.x) * 4;
  if (i < n) {
    float4 v = *(const float4*)(src + i);
    v4bf o; o[0]=(bf16_t)v.x; o[1]=(bf16_t)v.y; o[2]=(bf16_t)v.z; o[3]=(bf16_t)v.w;
    *(v4bf*)(dst + i) = o;
  }
}

// ---------- Kernel 3: GEMM  C[m,n] = sum_k A[m,k]*W[n,k] + bias[n] ----------
// A: M x 1024 bf16 row-major, W: 1024 x 1024 bf16 row-major (out,in).
// Block: 256 thr (8 waves). Tile M=256 (wave = 32 rows, 8 accumulators) x N=64.
// K chunk = 64, double-buffered via async global->LDS (ASYNCcnt).
// mode 0: write bf16 to (B,H,S,D). mode 1: f32 out + bias + residual.
#define WS_LD 80
__global__ __launch_bounds__(256)
void gemm_kernel(const bf16_t* __restrict__ A, const bf16_t* __restrict__ W,
                 const float* __restrict__ bias, bf16_t* __restrict__ out_bhsd,
                 float* __restrict__ out_f32, const float* __restrict__ residual,
                 int mode) {
  __shared__ bf16_t Ws[2][64][WS_LD];    // 2 x 64 rows(n) x 64 k (padded)
  const int K = E_DIM;
  int m0 = blockIdx.x * 256;
  int n0 = blockIdx.y * 64;
  int wid  = threadIdx.x >> 5;
  int lane = threadIdx.x & 31;
  int lm = lane & 15, lh = lane >> 4;
  int mrow = m0 + wid * 32;

  // staging map: 256 threads cover 64 rows x 64 cols (2 x 16B per thread)
  int srow = threadIdx.x >> 2;           // 0..63
  int scol = (threadIdx.x & 3) * 16;     // 0,16,32,48

  v8f c0[4] = {}, c1[4] = {};

  // prefetch chunk 0 into buffer 0
  {
    const bf16_t* g = W + (size_t)(n0 + srow) * K + scol;
    unsigned l = (unsigned)(uintptr_t)&Ws[0][srow][scol];
    async_copy16(l, g);
    async_copy16(l + 16, g + 8);
  }

  for (int i = 0; i < K / 64; ++i) {
    int buf = i & 1;
    int k0 = i * 64;
    __syncthreads();                      // everyone done reading buf^1
    if (i + 1 < K / 64) {                 // prefetch next chunk into buf^1
      const bf16_t* g = W + (size_t)(n0 + srow) * K + (k0 + 64) + scol;
      unsigned l = (unsigned)(uintptr_t)&Ws[buf ^ 1][srow][scol];
      async_copy16(l, g);
      async_copy16(l + 16, g + 8);
      wait_async_le2();                   // current chunk resident (in-order)
      __builtin_prefetch(A + (size_t)(mrow + lm) * K + k0 + 64, 0, 0);
    } else {
      wait_async_0();
    }
    __syncthreads();                      // all waves' copies complete

    #pragma unroll
    for (int kk = 0; kk < 64; kk += 32) {
      v16bf a0 = load_a_frag(A, mrow,      K, k0 + kk);
      v16bf a1 = load_a_frag(A, mrow + 16, K, k0 + kk);
      #pragma unroll
      for (int t = 0; t < 4; ++t) {
        v16bf b = load_b_frag(&Ws[buf][0][0], t * 16, WS_LD, kk);
        c0[t] = wmma_bf16f32(a0, b, c0[t]);
        c1[t] = wmma_bf16f32(a1, b, c1[t]);
      }
    }
  }

  // epilogue: C-frag lane holds column n = base + lm, rows r + lh*8
  #pragma unroll
  for (int t = 0; t < 4; ++t) {
    int n = n0 + t * 16 + lm;
    float bv = bias[n];
    #pragma unroll
    for (int half = 0; half < 2; ++half) {
      #pragma unroll
      for (int r = 0; r < 8; ++r) {
        int m = mrow + half * 16 + r + lh * 8;
        float val = (half ? c1[t][r] : c0[t][r]) + bv;
        if (mode == 0) {
          int b = m >> 11, s = m & (S_LEN - 1);
          int hh = n >> 6, d = n & (HD - 1);
          out_bhsd[(((size_t)(b * N_HEADS + hh) * S_LEN + s) << 6) + d] = (bf16_t)val;
        } else {
          size_t idx = (size_t)m * E_DIM + n;
          out_f32[idx] = val + residual[idx];
        }
      }
    }
  }
}

// ---------- Kernel 4: flash attention, one (b,h) x 128 queries per block ----------
#define KS_LD 80
#define VT_LD 48
#define PS_LD 48
__global__ __launch_bounds__(256)
void attn_kernel(const bf16_t* __restrict__ Q, const bf16_t* __restrict__ Km,
                 const bf16_t* __restrict__ V, bf16_t* __restrict__ Out) {
  __shared__ bf16_t Ks[32][KS_LD];        // 32 keys x 64 d, row-major
  __shared__ bf16_t Vt[64][VT_LD];        // transposed: d x 32 keys
  __shared__ bf16_t Ps[8][16][PS_LD];     // per-wave P tile 16 q x 32 keys

  int bh = blockIdx.x;                    // 0..63
  int b  = bh >> 4, h = bh & 15;
  const bf16_t* Qp = Q + (size_t)bh * S_LEN * HD;
  const bf16_t* Kp = Km + (size_t)bh * S_LEN * HD;
  const bf16_t* Vp = V + (size_t)bh * S_LEN * HD;

  int wid  = threadIdx.x >> 5;
  int lane = threadIdx.x & 31;
  int lm = lane & 15, lh = lane >> 4;
  int q0 = blockIdx.y * 128 + wid * 16;

  v16bf qa0 = load_a_frag(Qp, q0, HD, 0);     // d 0..31
  v16bf qa1 = load_a_frag(Qp, q0, HD, 32);    // d 32..63

  v8f o[4] = {};
  float mrow[8], lrow[8];
  #pragma unroll
  for (int r = 0; r < 8; ++r) { mrow[r] = -1e30f; lrow[r] = 0.0f; }

  int krow = threadIdx.x >> 3;            // 0..31
  int kcol = (threadIdx.x & 7) * 8;       // 0..56  (16B per thread)
  const float scale = 0.125f;             // 1/sqrt(64)

  for (int kc = 0; kc < S_LEN; kc += 32) {
    __syncthreads();
    // K chunk: async global->LDS straight copy
    async_copy16((unsigned)(uintptr_t)&Ks[krow][kcol],
                 Kp + (size_t)(kc + krow) * HD + kcol);
    // V chunk: register transpose into Vt (P@V B-frags need key-contiguous rows)
    v8bf vv = *(const v8bf*)(Vp + (size_t)(kc + krow) * HD + kcol);
    #pragma unroll
    for (int i = 0; i < 8; ++i) Vt[kcol + i][krow] = vv[i];
    wait_async_0();
    __syncthreads();

    // scores: S = Q @ K^T   (2 C-frags of 16 keys, inner d = 64 -> 2 k-steps)
    v8f s0 = {}, s1 = {};
    v16bf kb0 = load_b_frag(&Ks[0][0],  0, KS_LD, 0);
    v16bf kb1 = load_b_frag(&Ks[0][0], 16, KS_LD, 0);
    v16bf kb2 = load_b_frag(&Ks[0][0],  0, KS_LD, 32);
    v16bf kb3 = load_b_frag(&Ks[0][0], 16, KS_LD, 32);
    s0 = wmma_bf16f32(qa0, kb0, s0);
    s0 = wmma_bf16f32(qa1, kb2, s0);
    s1 = wmma_bf16f32(qa0, kb1, s1);
    s1 = wmma_bf16f32(qa1, kb3, s1);

    // online softmax: lane holds column lm of keys, rows r+lh*8
    float rmax[8], rsum[8], corr[8];
    #pragma unroll
    for (int r = 0; r < 8; ++r) {
      s0[r] *= scale; s1[r] *= scale;
      rmax[r] = fmaxf(s0[r], s1[r]);
    }
    #pragma unroll
    for (int mk = 1; mk <= 8; mk <<= 1)
      #pragma unroll
      for (int r = 0; r < 8; ++r)
        rmax[r] = fmaxf(rmax[r], __shfl_xor(rmax[r], mk, 32));
    #pragma unroll
    for (int r = 0; r < 8; ++r) {
      float mn = fmaxf(mrow[r], rmax[r]);
      corr[r] = __expf(mrow[r] - mn);
      mrow[r] = mn;
      s0[r] = __expf(s0[r] - mn);
      s1[r] = __expf(s1[r] - mn);
      rsum[r] = s0[r] + s1[r];
    }
    #pragma unroll
    for (int mk = 1; mk <= 8; mk <<= 1)
      #pragma unroll
      for (int r = 0; r < 8; ++r)
        rsum[r] += __shfl_xor(rsum[r], mk, 32);
    #pragma unroll
    for (int r = 0; r < 8; ++r) lrow[r] = lrow[r] * corr[r] + rsum[r];
    #pragma unroll
    for (int t = 0; t < 4; ++t)
      #pragma unroll
      for (int r = 0; r < 8; ++r) o[t][r] *= corr[r];

    // C-layout -> A-layout via wave-private LDS tile (lockstep wave32:
    // s_wait_dscnt 0 is sufficient, no block barrier needed)
    #pragma unroll
    for (int r = 0; r < 8; ++r) {
      int m = r + lh * 8;
      Ps[wid][m][lm]      = (bf16_t)s0[r];
      Ps[wid][m][16 + lm] = (bf16_t)s1[r];
    }
    wait_ds_0();

    // O += P @ V  (k = 32 keys, N = 64 d -> 4 WMMAs)
    v16bf pa = load_a_frag(&Ps[wid][0][0], 0, PS_LD, 0);
    #pragma unroll
    for (int t = 0; t < 4; ++t) {
      v16bf vb = load_b_frag(&Vt[0][0], t * 16, VT_LD, 0);
      o[t] = wmma_bf16f32(pa, vb, o[t]);
    }
  }

  // normalize and store to (B,S,E) bf16
  #pragma unroll
  for (int r = 0; r < 8; ++r) lrow[r] = 1.0f / lrow[r];
  #pragma unroll
  for (int t = 0; t < 4; ++t) {
    int d = t * 16 + lm;
    #pragma unroll
    for (int r = 0; r < 8; ++r) {
      int s = q0 + r + lh * 8;
      Out[((size_t)(b * S_LEN + s)) * E_DIM + h * HD + d] = (bf16_t)(o[t][r] * lrow[r]);
    }
  }
}

// ---------- host launch ----------
extern "C" void kernel_launch(void* const* d_in, const int* in_sizes, int n_in,
                              void* d_out, int out_size, void* d_ws, size_t ws_size,
                              hipStream_t stream) {
  const float* x   = (const float*)d_in[0];
  const float* Wq  = (const float*)d_in[1];
  const float* bq  = (const float*)d_in[2];
  const float* Wk  = (const float*)d_in[3];
  const float* bk  = (const float*)d_in[4];
  const float* Wv  = (const float*)d_in[5];
  const float* bv  = (const float*)d_in[6];
  const float* Wo  = (const float*)d_in[7];
  const float* bo  = (const float*)d_in[8];
  const float* gam = (const float*)d_in[9];
  const float* bet = (const float*)d_in[10];
  float* out = (float*)d_out;

  // workspace layout (bf16 elements): 88 MB total
  bf16_t* ws   = (bf16_t*)d_ws;
  const size_t ME = (size_t)M_TOK * E_DIM;   // 8M elems
  const size_t EE = (size_t)E_DIM * E_DIM;   // 1M elems
  bf16_t* hB    = ws;
  bf16_t* WqB   = hB   + ME;
  bf16_t* WkB   = WqB  + EE;
  bf16_t* WvB   = WkB  + EE;
  bf16_t* WoB   = WvB  + EE;
  bf16_t* Qb    = WoB  + EE;   // (B,H,S,D) = 8M elems
  bf16_t* Kb    = Qb   + ME;
  bf16_t* Vb    = Kb   + ME;
  bf16_t* attnB = Vb   + ME;   // (B,S,E)

  ln_cast_kernel<<<M_TOK, 256, 0, stream>>>(x, gam, bet, hB);
  int wblocks = (int)(EE / (256 * 4));
  cast_w_kernel<<<wblocks, 256, 0, stream>>>(Wq, WqB, (int)EE);
  cast_w_kernel<<<wblocks, 256, 0, stream>>>(Wk, WkB, (int)EE);
  cast_w_kernel<<<wblocks, 256, 0, stream>>>(Wv, WvB, (int)EE);
  cast_w_kernel<<<wblocks, 256, 0, stream>>>(Wo, WoB, (int)EE);

  dim3 ggrid(M_TOK / 256, E_DIM / 64);   // 32 x 16
  gemm_kernel<<<ggrid, 256, 0, stream>>>(hB, WqB, bq, Qb, nullptr, nullptr, 0);
  gemm_kernel<<<ggrid, 256, 0, stream>>>(hB, WkB, bk, Kb, nullptr, nullptr, 0);
  gemm_kernel<<<ggrid, 256, 0, stream>>>(hB, WvB, bv, Vb, nullptr, nullptr, 0);

  dim3 agrid(4 * N_HEADS, S_LEN / 128);  // 64 x 16
  attn_kernel<<<agrid, 256, 0, stream>>>(Qb, Kb, Vb, attnB);

  gemm_kernel<<<ggrid, 256, 0, stream>>>(attnB, WoB, bo, nullptr, out, x, 1);
}
